// BendingDiffSort_XY_9792525435031
// MI455X (gfx1250) — matrix operand order repaired and checked by reading, not compiled, not run on hardware
//
#include <hip/hip_runtime.h>
#include <hip/hip_bf16.h>

// ---------------------------------------------------------------------------
// Problem constants (from reference)
// ---------------------------------------------------------------------------
#define BB   32
#define CC   128
#define HID  256
#define NN   64
#define NP   4096   // N*N pixels per (b,c) slice

typedef __bf16 bf16;
typedef bf16  v16bf __attribute__((ext_vector_type(16)));
typedef bf16  v8bf  __attribute__((ext_vector_type(8)));
typedef float v8f   __attribute__((ext_vector_type(8)));

__device__ inline bf16 f2bf(float f) {
  // round-to-nearest-even f32 -> bf16
  unsigned int u = __float_as_uint(f);
  u += 0x7fffu + ((u >> 16) & 1u);
  unsigned short s = (unsigned short)(u >> 16);
  union { unsigned short u; bf16 b; } cvt; cvt.u = s;
  return cvt.b;
}

// Unused helper (A is staged unpermuted): K groups of 8, group g (k = 8g+j) goes to
// position-group gs: 0->0, 1->2, 2->1, 3->3 (as staged, lane<16 reads pos 0..15 =
// K0..15; lane>=16 reads pos 16..31 = K16..31).
__device__ inline int gswap(int g) { return ((g & 1) << 1) | (g >> 1); }

// ---------------------------------------------------------------------------
// Kernel 0: f32 -> bf16 convert, 8 elements per thread (b128 in/out)
// ---------------------------------------------------------------------------
__global__ __launch_bounds__(256) void cvt_f32_bf16(const float* __restrict__ in,
                                                    bf16* __restrict__ out, int n8) {
  int i = blockIdx.x * 256 + threadIdx.x;
  if (i < n8) {
    const float4* p = (const float4*)(in + (size_t)i * 8);
    float4 a = p[0], b = p[1];
    v8bf v;
    v[0] = f2bf(a.x); v[1] = f2bf(a.y); v[2] = f2bf(a.z); v[3] = f2bf(a.w);
    v[4] = f2bf(b.x); v[5] = f2bf(b.y); v[6] = f2bf(b.z); v[7] = f2bf(b.w);
    *(v8bf*)(out + (size_t)i * 8) = v;
  }
}

// ---------------------------------------------------------------------------
// Kernel 1: H = relu(W(HIDxK) * X(KxNP) + bias), per batch, bf16 WMMA.
// 256 threads = 8 waves (2M x 4N), each wave owns a 2x2 grid of 16x16 tiles
// -> 64(M) x 128(N) block tile.  K staged in chunks of 64 (two 32-sub-chunks),
// 8 WMMAs per wave per stage.  All staging is b128 global loads; A goes to LDS
// as b128 stores (k contiguous, unpermuted), B is transposed to n-major with
// k-pair zipped b32 stores.  Every WMMA operand is one 32B v16bf LDS load.
// ---------------------------------------------------------------------------
template<int K>
__global__ __launch_bounds__(256) void gemm_bias_relu(
    const bf16* __restrict__ W,    // [HID][K] bf16
    const bf16* __restrict__ X,    // [B][K][NP] bf16
    const float* __restrict__ bias,// [HID]
    bf16* __restrict__ Out) {      // [B][HID][NP] bf16
  const int n0 = blockIdx.x * 128;
  const int m0 = blockIdx.y * 64;
  const int b  = blockIdx.z;

  __shared__ alignas(32) bf16 sA[64][64];    // [m][k] (k contiguous)    8 KB
  __shared__ alignas(32) bf16 sBt[128][64];  // [n][k] (n-major)        16 KB

  const int tid  = threadIdx.x;
  const int lane = tid & 31;
  const int wave = tid >> 5;
  const int wm = wave >> 2;     // 0..1 : M sub-panels of 32 rows
  const int wn = wave & 3;      // 0..3 : N sub-panels of 32 cols
  const int l15  = lane & 15;
  const int half = (lane < 16) ? 0 : 16;

  v8f acc[2][2] = {};

  for (int kc = 0; kc < K; kc += 64) {
    // ---- load stage (all global b128 loads issued before LDS stores) ----
    v8bf aw[2], xr0[2], xr1[2];
#pragma unroll
    for (int cc = 0; cc < 2; ++cc) {
      const int ca = tid + cc * 256;          // 0..511
      const int r  = ca >> 3, g6 = ca & 7;    // A: row r, k-group g6
      aw[cc] = *(const v8bf*)&W[(size_t)(m0 + r) * K + kc + g6 * 8];
      const int kp = ca >> 4, ng = ca & 15;   // B: k-pair kp, n-group ng
      const bf16* xp = &X[((size_t)b * K + kc + 2 * kp) * NP + n0 + ng * 8];
      xr0[cc] = *(const v8bf*)xp;
      xr1[cc] = *(const v8bf*)(xp + NP);
    }
#pragma unroll
    for (int cc = 0; cc < 2; ++cc) {
      const int ca = tid + cc * 256;
      const int r  = ca >> 3, g6 = ca & 7;
      const int sub = g6 >> 2, g = g6 & 3;
      *(v8bf*)&sA[r][sub * 32 + g * 8] = aw[cc];
      const int kp = ca >> 4, ng = ca & 15;
#pragma unroll
      for (int j = 0; j < 8; ++j) {
        union { bf16 b2[2]; unsigned int u; } pk;
        pk.b2[0] = xr0[cc][j]; pk.b2[1] = xr1[cc][j];
        *(unsigned int*)&sBt[ng * 8 + j][2 * kp] = pk.u;
      }
    }
    // prefetch next K panel of X (global_prefetch_b8)
    if (kc + 64 < K)
      __builtin_prefetch(
          (const void*)&X[((size_t)b * K + kc + 64 + (tid & 63)) * NP + n0 + (tid >> 6) * 32],
          0, 1);
    __syncthreads();

    // ---- compute: 2 sub-chunks x (2x2 tiles) = 8 WMMAs ----
#pragma unroll
    for (int sub = 0; sub < 2; ++sub) {
      v16bf af[2], bfr[2];
#pragma unroll
      for (int i = 0; i < 2; ++i)
        af[i] = *(const v16bf*)&sA[wm * 32 + i * 16 + l15][sub * 32 + half];
#pragma unroll
      for (int j = 0; j < 2; ++j)
        bfr[j] = *(const v16bf*)&sBt[wn * 32 + j * 16 + l15][sub * 32 + half];
#pragma unroll
      for (int i = 0; i < 2; ++i)
#pragma unroll
        for (int j = 0; j < 2; ++j)
          acc[i][j] = __builtin_amdgcn_wmma_f32_16x16x32_bf16(
              false, af[i], false, bfr[j], (short)0, acc[i][j], false, false);
    }
    __syncthreads();
  }

  // D layout: lane<16 -> rows r, lane>=16 -> rows r+8; col = lane%16
#pragma unroll
  for (int i = 0; i < 2; ++i) {
    int mbase = m0 + wm * 32 + i * 16 + ((lane < 16) ? 0 : 8);
#pragma unroll
    for (int j = 0; j < 2; ++j) {
      int ncol = n0 + wn * 32 + j * 16 + l15;
#pragma unroll
      for (int r = 0; r < 8; ++r) {
        int mrow = mbase + r;
        float v = fmaxf(acc[i][j][r] + bias[mrow], 0.0f);
        Out[((size_t)b * HID + mrow) * NP + ncol] = f2bf(v);
      }
    }
  }
}

// ---------------------------------------------------------------------------
// Kernel 2: row/col score reductions over h2 (bf16) with f32 accumulation.
// scores layout: [2][B][64]  (0 = row-scores over w, 1 = col-scores over h)
// ---------------------------------------------------------------------------
__global__ __launch_bounds__(256) void score_reduce(
    const bf16* __restrict__ H,       // [B][HID][64][64]
    const float* __restrict__ w_row,  // [HID][64]
    const float* __restrict__ b_row,  // [1]
    const float* __restrict__ w_col,  // [HID][64]
    const float* __restrict__ b_col,  // [1]
    float* __restrict__ scores) {
  const int idx   = blockIdx.x;  // 0..63
  const int which = blockIdx.y;  // 0 row, 1 col
  const int b     = blockIdx.z;
  const int tid   = threadIdx.x;

  float part = 0.f;
  if (which == 0) {
    for (int t = tid; t < HID * 64; t += 256) {
      int c = t >> 6, w = t & 63;
      part += (float)H[(((size_t)b * HID + c) * 64 + idx) * 64 + w] * w_row[c * 64 + w];
    }
  } else {
    for (int t = tid; t < HID * 64; t += 256) {
      int c = t >> 6, h = t & 63;
      part += (float)H[(((size_t)b * HID + c) * 64 + h) * 64 + idx] * w_col[c * 64 + h];
    }
  }
  __shared__ float red[256];
  red[tid] = part;
  __syncthreads();
  for (int s = 128; s > 0; s >>= 1) {
    if (tid < s) red[tid] += red[tid + s];
    __syncthreads();
  }
  if (tid == 0) {
    float bias = (which == 0) ? b_row[0] : b_col[0];
    scores[((size_t)which * BB + b) * 64 + idx] = red[0] + bias;
  }
}

// ---------------------------------------------------------------------------
// Kernel 3: differentiable bitonic sort (Cauchy CDF), P held in LDS.
// One block per (axis, batch). Emits soft permutation P as bf16 [2][B][64][64].
// ---------------------------------------------------------------------------
__global__ __launch_bounds__(256) void diffsort_kernel(
    const float* __restrict__ scores,  // [2][B][64]
    bf16* __restrict__ Pout) {         // [2][B][64][64]
  const int which = blockIdx.x;
  const int b     = blockIdx.y;
  const int tid   = threadIdx.x;

  __shared__ float sx[64];
  __shared__ float sP[64][65];
  __shared__ float sal[32];
  __shared__ int   sai[32], sbi[32];

  if (tid < 64) sx[tid] = scores[((size_t)which * BB + b) * 64 + tid];
  for (int t = tid; t < 64 * 64; t += 256)
    sP[t >> 6][t & 63] = ((t >> 6) == (t & 63)) ? 1.f : 0.f;
  __syncthreads();

  for (int blk = 0; blk < 6; ++blk) {
    for (int lay = 0; lay <= blk; ++lay) {
      const int m = 1 << (blk - lay);
      if (tid < 32) {
        int j  = tid & (m - 1);
        int i  = (tid / m) * (2 * m);
        int ix = i + j;
        int ai = ix, bi = ix + m;
        if ((ix >> (blk + 1)) & 1) { int tmp = ai; ai = bi; bi = tmp; }
        float xa = sx[ai], xb = sx[bi];
        float alpha = atanf((xb - xa) * 50.0f) * 0.3183098861837907f + 0.5f;
        sal[tid] = alpha; sai[tid] = ai; sbi[tid] = bi;
        sx[ai] = alpha * xa + (1.f - alpha) * xb;
        sx[bi] = (1.f - alpha) * xa + alpha * xb;
      }
      __syncthreads();
      // update P columns: 32 pairs x 64 rows
      for (int t = tid; t < 2048; t += 256) {
        int p = t >> 6, r = t & 63;
        float al = sal[p];
        int ai = sai[p], bi = sbi[p];
        float Pa = sP[r][ai], Pb = sP[r][bi];
        sP[r][ai] = al * Pa + (1.f - al) * Pb;
        sP[r][bi] = (1.f - al) * Pa + al * Pb;
      }
      __syncthreads();
    }
  }
  for (int t = tid; t < 4096; t += 256)
    Pout[((size_t)which * BB + b) * 4096 + t] = f2bf(sP[t >> 6][t & 63]);
}

// ---------------------------------------------------------------------------
// Kernel 4: apply soft permutation.  512 threads = 16 waves, one 16x16 WMMA
// tile each, K = 64 in two sub-chunks.  A staged fragment-ordered (b128),
// B staged n-major.
//  TRANSB=false: y   = P_row @ x    (B transposed while staging; bf16 out)
//  TRANSB=true : out = P_col @ y^T  (row-major y IS the transposed B; f32 out)
// ---------------------------------------------------------------------------
template<bool TRANSB>
__global__ __launch_bounds__(512) void perm_mm(
    const bf16* __restrict__ P,    // [B][64][64]
    const bf16* __restrict__ Xin,  // [B][C][64][64]
    bf16* __restrict__ Ybf,        // bf16 out (TRANSB=false)
    float* __restrict__ Fout) {    // f32 out  (TRANSB=true)
  const int c = blockIdx.x;
  const int b = blockIdx.y;
  const int tid  = threadIdx.x;
  const int lane = tid & 31;
  const int wave = tid >> 5;
  const int tm = wave >> 2;   // 0..3
  const int tn = wave & 3;    // 0..3
  const int l15  = lane & 15;
  const int half = (lane < 16) ? 0 : 16;

  __shared__ alignas(32) bf16 sA[64][64];   // [m][k] (k contiguous)
  __shared__ alignas(32) bf16 sB[64][64];   // [n][k] (n-major)

  // A: one group-of-8 chunk per thread (512 chunks)
  {
    const int r = tid >> 3, g6 = tid & 7;
    const int sub = g6 >> 2, g = g6 & 3;
    v8bf v = *(const v8bf*)&P[(size_t)b * 4096 + r * 64 + g6 * 8];
    *(v8bf*)&sA[r][sub * 32 + g * 8] = v;
  }
  const size_t xbase = ((size_t)b * CC + c) * 4096;
  if (TRANSB) {
    // straight row-major copy (b128 both sides)
    const int r = tid >> 3, g = tid & 7;
    v8bf v = *(const v8bf*)&Xin[xbase + r * 64 + g * 8];
    *(v8bf*)&sB[r][g * 8] = v;
  } else {
    // transpose while staging: k-pair zip, b32 LDS stores
    if (tid < 256) {
      const int kp = tid >> 3, ng = tid & 7;
      const bf16* xp = &Xin[xbase + (size_t)(2 * kp) * 64 + ng * 8];
      v8bf r0 = *(const v8bf*)xp;
      v8bf r1 = *(const v8bf*)(xp + 64);
#pragma unroll
      for (int j = 0; j < 8; ++j) {
        union { bf16 b2[2]; unsigned int u; } pk;
        pk.b2[0] = r0[j]; pk.b2[1] = r1[j];
        *(unsigned int*)&sB[ng * 8 + j][2 * kp] = pk.u;
      }
    }
  }
  __syncthreads();

  v8f acc = {};
#pragma unroll
  for (int sub = 0; sub < 2; ++sub) {
    v16bf a  = *(const v16bf*)&sA[tm * 16 + l15][sub * 32 + half];
    v16bf bm = *(const v16bf*)&sB[tn * 16 + l15][sub * 32 + half];
    acc = __builtin_amdgcn_wmma_f32_16x16x32_bf16(false, a, false, bm,
                                                  (short)0, acc, false, false);
  }

  const int ncol  = tn * 16 + l15;
  const int mbase = tm * 16 + ((lane < 16) ? 0 : 8);
#pragma unroll
  for (int r = 0; r < 8; ++r) {
    int mrow = mbase + r;
    if (TRANSB) Fout[xbase + (size_t)mrow * 64 + ncol] = acc[r];
    else        Ybf [xbase + (size_t)mrow * 64 + ncol] = f2bf(acc[r]);
  }
}

// ---------------------------------------------------------------------------
// Host launcher
// ---------------------------------------------------------------------------
extern "C" void kernel_launch(void* const* d_in, const int* in_sizes, int n_in,
                              void* d_out, int out_size, void* d_ws, size_t ws_size,
                              hipStream_t stream) {
  const float* x     = (const float*)d_in[0];
  const float* w1    = (const float*)d_in[1];
  const float* b1    = (const float*)d_in[2];
  const float* w2    = (const float*)d_in[3];
  const float* b2    = (const float*)d_in[4];
  const float* w_row = (const float*)d_in[5];
  const float* b_row = (const float*)d_in[6];
  const float* w_col = (const float*)d_in[7];
  const float* b_col = (const float*)d_in[8];
  float* out = (float*)d_out;

  // workspace carve-out (256B aligned)
  char* ws = (char*)d_ws;
  size_t off = 0;
  auto carve = [&](size_t bytes) -> void* {
    void* p = ws + off;
    off += (bytes + 255) & ~(size_t)255;
    return p;
  };
  bf16* x_bf   = (bf16*)carve((size_t)BB * CC * NP * 2);   //  33.5 MB
  bf16* w1_bf  = (bf16*)carve((size_t)HID * CC * 2);
  bf16* w2_bf  = (bf16*)carve((size_t)HID * HID * 2);
  bf16* h1_bf  = (bf16*)carve((size_t)BB * HID * NP * 2);  //  67 MB
  bf16* h2_bf  = (bf16*)carve((size_t)BB * HID * NP * 2);  //  67 MB
  float* scores = (float*)carve((size_t)2 * BB * 64 * 4);
  bf16* P_bf   = (bf16*)carve((size_t)2 * BB * 4096 * 2);  // [2][B][64][64]
  bf16* y_bf   = (bf16*)carve((size_t)BB * CC * NP * 2);   //  33.5 MB
  (void)ws_size; (void)in_sizes; (void)n_in; (void)out_size;

  // 0. converts (8 elements per thread)
  {
    int n8 = (BB * CC * NP) / 8;
    cvt_f32_bf16<<<(n8 + 255) / 256, 256, 0, stream>>>(x, x_bf, n8);
    n8 = (HID * CC) / 8;
    cvt_f32_bf16<<<(n8 + 255) / 256, 256, 0, stream>>>(w1, w1_bf, n8);
    n8 = (HID * HID) / 8;
    cvt_f32_bf16<<<(n8 + 255) / 256, 256, 0, stream>>>(w2, w2_bf, n8);
  }

  // 1. h1 = relu(w1 @ x + b1)   : K=128  (block tile 64x128)
  gemm_bias_relu<CC><<<dim3(NP / 128, HID / 64, BB), 256, 0, stream>>>(
      w1_bf, x_bf, b1, h1_bf);

  // 2. h2 = relu(w2 @ h1 + b2)  : K=256
  gemm_bias_relu<HID><<<dim3(NP / 128, HID / 64, BB), 256, 0, stream>>>(
      w2_bf, h1_bf, b2, h2_bf);

  // 3. row/col scores
  score_reduce<<<dim3(64, 2, BB), 256, 0, stream>>>(
      h2_bf, w_row, b_row, w_col, b_col, scores);

  // 4. diffsort -> soft permutation matrices (bf16)
  diffsort_kernel<<<dim3(2, BB), 256, 0, stream>>>(scores, P_bf);

  // 5a. y = P_row @ x
  perm_mm<false><<<dim3(CC, BB), 512, 0, stream>>>(
      P_bf /* which=0 */, x_bf, y_bf, nullptr);

  // 5b. out = P_col @ y^T  (f32 output)
  perm_mm<true><<<dim3(CC, BB), 512, 0, stream>>>(
      P_bf + (size_t)BB * 4096 /* which=1 */, y_bf, nullptr, out);
}
